// TopicVectorQuantizedVAE_64613488001057
// MI455X (gfx1250) — matrix operand order repaired and checked by reading, not compiled, not run on hardware
//
#include <hip/hip_runtime.h>

typedef __attribute__((ext_vector_type(16))) _Float16 v16h;
typedef __attribute__((ext_vector_type(8)))  _Float16 v8h;
typedef __attribute__((ext_vector_type(8)))  float    v8f;

#define GT_M 64
#define GT_N 64
#define CHUNK_ROWS 4096

// ---------------------------------------------------------------------------
// Layout transforms
// ---------------------------------------------------------------------------
__global__ void k_nchw2nhwc(const float* __restrict__ in, float* __restrict__ out,
                            int Bb, int C, int Hh, int Ww) {
    long total = (long)Bb * C * Hh * Ww;
    for (long i = blockIdx.x * (long)blockDim.x + threadIdx.x; i < total;
         i += (long)gridDim.x * blockDim.x) {
        int c = (int)(i % C);
        long t = i / C;
        int x = (int)(t % Ww); t /= Ww;
        int y = (int)(t % Hh); t /= Hh;
        int b = (int)t;
        out[i] = in[(((long)b * C + c) * Hh + y) * Ww + x];
    }
}

__global__ void k_nhwc2nchw(const float* __restrict__ in, float* __restrict__ out,
                            int Bb, int C, int Hh, int Ww) {
    long total = (long)Bb * C * Hh * Ww;
    for (long i = blockIdx.x * (long)blockDim.x + threadIdx.x; i < total;
         i += (long)gridDim.x * blockDim.x) {
        int x = (int)(i % Ww);
        long t = i / Ww;
        int y = (int)(t % Hh); t /= Hh;
        int c = (int)(t % C);  t /= C;
        int b = (int)t;
        out[i] = in[(((long)b * Hh + y) * Ww + x) * C + c];
    }
}

// ---------------------------------------------------------------------------
// im2col: NHWC f32 -> (Mc x Kp) f16 patch matrix. Supports lhs dilation (for
// transposed conv lowered to regular conv) and fused pre-ReLU.
// Patch element order: c*k*k + ky*k + kx. Rows >= C*k*k are zero padding.
// ---------------------------------------------------------------------------
__global__ void k_im2col(const float* __restrict__ src, _Float16* __restrict__ dst,
                         int Cc, int Hh, int Ww, int OH, int OW,
                         int ksz, int stride, int pad, int dil, int Kp,
                         int m0, int Mc, int relu) {
    long total = (long)Mc * Kp;
    int kk = ksz * ksz;
    for (long i = blockIdx.x * (long)blockDim.x + threadIdx.x; i < total;
         i += (long)gridDim.x * blockDim.x) {
        int kidx = (int)(i % Kp);
        int mi   = (int)(i / Kp);
        int m = m0 + mi;
        int ox = m % OW; int t = m / OW;
        int oy = t % OH; int b = t / OH;
        float v = 0.f;
        int c = kidx / kk;
        if (c < Cc) {
            int rem = kidx - c * kk;
            int ky = rem / ksz, kx = rem % ksz;
            int ay = oy * stride - pad + ky;
            int ax = ox * stride - pad + kx;
            bool ok = (ay >= 0) && (ax >= 0);
            if (ok && dil > 1) {
                ok = ((ay % dil) == 0) && ((ax % dil) == 0);
                ay /= dil; ax /= dil;
            }
            ok = ok && (ay < Hh) && (ax < Ww);
            if (ok) {
                v = src[(((long)b * Hh + ay) * Ww + ax) * Cc + c];
                if (relu && v < 0.f) v = 0.f;
            }
        }
        dst[(long)mi * Kp + kidx] = (_Float16)v;
    }
}

// ---------------------------------------------------------------------------
// Weight prep: f32 (O,I,k,k) [or transposed-conv (I,O,k,k) with spatial flip]
// -> f16 stored TRANSPOSED as (O x Kp) row-major, i.e. B^T (column-major B),
// so GEMM B-tile loads are contiguous. Row index within Kp: i*k*k + ky*k + kx.
// ---------------------------------------------------------------------------
__global__ void k_prep_w(const float* __restrict__ src, _Float16* __restrict__ dst,
                         int O, int I, int ksz, int Kp, int tmode) {
    long total = (long)Kp * O;
    int kk = ksz * ksz;
    for (long i = blockIdx.x * (long)blockDim.x + threadIdx.x; i < total;
         i += (long)gridDim.x * blockDim.x) {
        int kidx = (int)(i % Kp);
        int o = (int)(i / Kp);
        float v = 0.f;
        int c = kidx / kk;
        if (c < I) {
            int rem = kidx - c * kk;
            int ky = rem / ksz, kx = rem % ksz;
            if (tmode)
                v = src[(((long)c * O + o) * ksz + (ksz - 1 - ky)) * ksz + (ksz - 1 - kx)];
            else
                v = src[(((long)o * I + c) * ksz + ky) * ksz + kx];
        }
        dst[i] = (_Float16)v;   // dst[o*Kp + kidx]
    }
}

// ---------------------------------------------------------------------------
// WMMA GEMM: C(Mc x N) = A(Mc x Kp, f16) * B(Kp x N, f16 given as B^T NxKp)
// + bias, f32 out. 128 threads = 4 waves; 64x64 tile; each wave 32x32
// (2x2 v_wmma_f32_16x16x32_f16 tiles). Register-prefetch pipeline over K.
// ---------------------------------------------------------------------------
__launch_bounds__(128)
__global__ void k_gemm_wmma(const _Float16* __restrict__ A, const _Float16* __restrict__ Bt,
                            float* __restrict__ C, const float* __restrict__ bias,
                            int N, int Kp, int relu) {
    __shared__ alignas(64) _Float16 sA[GT_M * 32];   // 64 rows x 32 k
    __shared__ alignas(64) _Float16 sB[GT_N * 32];   // 64 cols x 32 k (B^T)
    int bm = blockIdx.x;
    int bn = blockIdx.y;
    int tid = threadIdx.x;
    int lane = tid & 31;
    int w = tid >> 5;
    int wm = w >> 1, wn = w & 1;

    v8f acc[2][2];
    for (int i = 0; i < 2; i++)
        for (int j = 0; j < 2; j++)
            for (int r = 0; r < 8; r++) acc[i][j][r] = 0.f;

    int arow = lane & 15;
    int kA0 = (lane >> 4) * 16;   // A frag: lane<16 -> K 0..15, else K 16..31
    int kB0 = (lane >> 4) * 16;   // B frag: lane<16 -> K 0..15, else K 16..31

    // cooperative tile staging: thread t handles row/col rr, 16-half chunk cc
    int rr = tid >> 1;
    int cc = (tid & 1) * 16;
    const _Float16* gA = &A [((long)(bm * GT_M + rr)) * Kp + cc];
    const _Float16* gB = &Bt[((long)(bn * GT_N + rr)) * Kp + cc];
    uint4* sa = (uint4*)&sA[rr * 32 + cc];
    uint4* sb = (uint4*)&sB[rr * 32 + cc];

    uint4 ra0, ra1, rb0, rb1;
    {
        const uint4* qa = (const uint4*)gA;
        const uint4* qb = (const uint4*)gB;
        ra0 = qa[0]; ra1 = qa[1];
        rb0 = qb[0]; rb1 = qb[1];
    }

    for (int k0 = 0; k0 < Kp; k0 += 32) {
        sa[0] = ra0; sa[1] = ra1;
        sb[0] = rb0; sb[1] = rb1;
        __syncthreads();

        if (k0 + 32 < Kp) {   // prefetch next chunk while WMMAs run
            const uint4* qa = (const uint4*)(gA + k0 + 32);
            const uint4* qb = (const uint4*)(gB + k0 + 32);
            ra0 = qa[0]; ra1 = qa[1];
            rb0 = qb[0]; rb1 = qb[1];
        }

        v16h af[2], bf[2];
#pragma unroll
        for (int i = 0; i < 2; i++) {
            const _Float16* base = &sA[(wm * 32 + i * 16 + arow) * 32];
            v8h lo = *(const v8h*)&base[kA0];
            v8h hi = *(const v8h*)&base[kA0 + 8];
            af[i] = __builtin_shufflevector(lo, hi, 0, 1, 2, 3, 4, 5, 6, 7,
                                            8, 9, 10, 11, 12, 13, 14, 15);
        }
#pragma unroll
        for (int j = 0; j < 2; j++) {
            const _Float16* base = &sB[(wn * 32 + j * 16 + arow) * 32];
            bf[j] = *(const v16h*)&base[kB0];
        }
#pragma unroll
        for (int i = 0; i < 2; i++)
#pragma unroll
            for (int j = 0; j < 2; j++)
                acc[i][j] = __builtin_amdgcn_wmma_f32_16x16x32_f16(
                    false, af[i], false, bf[j], (short)0, acc[i][j], false, false);
        __syncthreads();
    }

    int col = lane & 15;
    int row0 = (lane >> 4) * 8;   // C/D: vgpr r -> M = row0 + r
    for (int i = 0; i < 2; i++)
        for (int j = 0; j < 2; j++) {
            int gr = bm * GT_M + wm * 32 + i * 16 + row0;
            int gc = bn * GT_N + wn * 32 + j * 16 + col;
            float bv = bias ? bias[gc] : 0.f;
#pragma unroll
            for (int r = 0; r < 8; r++) {
                float v = acc[i][j][r] + bv;
                if (relu && v < 0.f) v = 0.f;
                C[((long)(gr + r)) * N + gc] = v;
            }
        }
}

// ---------------------------------------------------------------------------
// BatchNorm2d over NHWC rows (channels = 256 = blockDim for stage1).
// ---------------------------------------------------------------------------
__global__ void k_bnstats1(const float* __restrict__ in, float* __restrict__ part,
                           int rowsPerBlk, int C) {
    int ch = threadIdx.x;
    long rbase = (long)blockIdx.x * rowsPerBlk;
    float s = 0.f, q = 0.f;
    for (int r = 0; r < rowsPerBlk; r++) {
        float v = in[(rbase + r) * C + ch];
        s += v; q += v * v;
    }
    part[(long)blockIdx.x * C + ch] = s;
    part[((long)gridDim.x + blockIdx.x) * C + ch] = q;
}

__global__ void k_bnstats2(const float* __restrict__ part, const float* __restrict__ g,
                           const float* __restrict__ b, float* __restrict__ ss,
                           int nb, int C, int M) {
    int ch = threadIdx.x;
    float s = 0.f, q = 0.f;
    for (int i = 0; i < nb; i++) {
        s += part[(long)i * C + ch];
        q += part[((long)nb + i) * C + ch];
    }
    float mean = s / (float)M;
    float var  = q / (float)M - mean * mean;
    float sc = rsqrtf(var + 1e-5f);
    if (g) sc *= g[ch];
    float sh = -mean * sc;
    if (b) sh += b[ch];
    ss[ch] = sc;
    ss[C + ch] = sh;
}

__global__ void k_bnapply(const float* __restrict__ in, const float* __restrict__ ss,
                          const float* __restrict__ resid, float* __restrict__ out,
                          long total, int C, int relu) {
    for (long i = blockIdx.x * (long)blockDim.x + threadIdx.x; i < total;
         i += (long)gridDim.x * blockDim.x) {
        int c = (int)(i % C);
        float v = in[i] * ss[c] + ss[C + c];
        if (resid) v += resid[i];
        else if (relu && v < 0.f) v = 0.f;
        out[i] = v;
    }
}

// ---------------------------------------------------------------------------
// VQ: exact f32 nearest-neighbor (codebook resident in L2), then gather.
// ---------------------------------------------------------------------------
__global__ void k_vq_argmin(const float* __restrict__ zf, const float* __restrict__ E,
                            int* __restrict__ idx, int Dd, int Kk) {
    __shared__ float z[256];
    __shared__ float bd[256];
    __shared__ int   bi[256];
    int m = blockIdx.x, t = threadIdx.x;
    z[t] = zf[(long)m * Dd + t];
    __syncthreads();
    float best = 3.4e38f; int bj = 0;
    for (int j = t; j < Kk; j += 256) {
        const float* e = &E[(long)j * Dd];
        float d = 0.f;
        for (int c = 0; c < Dd; c++) { float df = z[c] - e[c]; d += df * df; }
        if (d < best || (d == best && j < bj)) { best = d; bj = j; }
    }
    bd[t] = best; bi[t] = bj;
    __syncthreads();
    for (int s = 128; s > 0; s >>= 1) {
        if (t < s) {
            float d2 = bd[t + s]; int i2 = bi[t + s];
            if (d2 < bd[t] || (d2 == bd[t] && i2 < bi[t])) { bd[t] = d2; bi[t] = i2; }
        }
        __syncthreads();
    }
    if (t == 0) idx[m] = bi[0];
}

__global__ void k_gather(const int* __restrict__ idx, const float* __restrict__ E,
                         float* __restrict__ out, long total, int Dd) {
    for (long i = blockIdx.x * (long)blockDim.x + threadIdx.x; i < total;
         i += (long)gridDim.x * blockDim.x) {
        long m = i / Dd; int c = (int)(i % Dd);
        out[i] = E[(long)idx[m] * Dd + c];
    }
}

// ---------------------------------------------------------------------------
// ConvTranspose2d 256->3 (k4,s2,p1) + tanh, writes NCHW output directly.
// ---------------------------------------------------------------------------
__global__ void k_convT2(const float* __restrict__ in, const float* __restrict__ wgt,
                         const float* __restrict__ bias, float* __restrict__ out) {
    int i = blockIdx.x * blockDim.x + threadIdx.x;   // over B*32*32
    if (i >= 256 * 32 * 32) return;
    int x = i & 31, y = (i >> 5) & 31, b = i >> 10;
    float a0 = bias[0], a1 = bias[1], a2 = bias[2];
    for (int ky = 0; ky < 4; ky++) {
        int ay = y + 1 - ky;
        if (ay & 1) continue;
        int iy = ay >> 1;
        if (iy < 0 || iy >= 16) continue;
        for (int kx = 0; kx < 4; kx++) {
            int ax = x + 1 - kx;
            if (ax & 1) continue;
            int ix = ax >> 1;
            if (ix < 0 || ix >= 16) continue;
            const float* ip = &in[(((long)b * 16 + iy) * 16 + ix) * 256];
            for (int ci = 0; ci < 256; ci++) {
                float v = ip[ci];
                long wb = ((long)ci * 3) * 16 + ky * 4 + kx;
                a0 += v * wgt[wb];
                a1 += v * wgt[wb + 16];
                a2 += v * wgt[wb + 32];
            }
        }
    }
    long ob = ((long)b * 3) * 1024 + y * 32 + x;
    out[ob]        = tanhf(a0);
    out[ob + 1024] = tanhf(a1);
    out[ob + 2048] = tanhf(a2);
}

// ---------------------------------------------------------------------------
// Topic head
// ---------------------------------------------------------------------------
__global__ void k_topic_t0(const int* __restrict__ idx, const float* __restrict__ infw,
                           const float* __restrict__ infb, float* __restrict__ t) {
    __shared__ int id[64];
    int b = blockIdx.x, tt = threadIdx.x;
    if (tt < 64) id[tt] = idx[b * 64 + tt];
    __syncthreads();
    if (tt < 100) {
        float acc = infb[tt];
        for (int p = 0; p < 64; p++)
            acc += infw[((long)tt * 512 + id[p]) * 64 + p];
        t[(long)b * 100 + tt] = tanhf(acc);
    }
}

__global__ void k_linear(const float* __restrict__ in, const float* __restrict__ W,
                         const float* __restrict__ bias, float* __restrict__ out,
                         int J, int F, int wtrans, int do_tanh) {
    __shared__ float row[512];
    int b = blockIdx.x, t = threadIdx.x;
    for (int j = t; j < J; j += blockDim.x) row[j] = in[(long)b * J + j];
    __syncthreads();
    for (int f = t; f < F; f += blockDim.x) {
        float acc = bias ? bias[f] : 0.f;
        if (wtrans) for (int j = 0; j < J; j++) acc += row[j] * W[(long)j * F + f];
        else        for (int j = 0; j < J; j++) acc += row[j] * W[(long)f * J + j];
        if (do_tanh) acc = tanhf(acc);
        out[(long)b * F + f] = acc;
    }
}

__global__ void k_bn1d(const float* __restrict__ in, float* __restrict__ out, int F) {
    __shared__ float s1[256], s2[256];
    int f = blockIdx.x, r = threadIdx.x;       // blockDim == B == 256
    float v = in[(long)r * F + f];
    s1[r] = v; s2[r] = v * v;
    __syncthreads();
    for (int s = 128; s > 0; s >>= 1) {
        if (r < s) { s1[r] += s1[r + s]; s2[r] += s2[r + s]; }
        __syncthreads();
    }
    float mean = s1[0] / 256.f;
    float var  = s2[0] / 256.f - mean * mean;
    out[(long)r * F + f] = (v - mean) * rsqrtf(var + 1e-5f);
}

__global__ void k_theta(const float* __restrict__ mu, const float* __restrict__ lv,
                        const float* __restrict__ eps, float* __restrict__ theta) {
    int b = blockIdx.x;
    if (threadIdx.x != 0) return;
    float v[10];
    float mx = -3.4e38f;
    for (int c = 0; c < 10; c++) {
        v[c] = mu[b * 10 + c] + eps[b * 10 + c] * expf(0.5f * lv[b * 10 + c]);
        mx = fmaxf(mx, v[c]);
    }
    float s = 0.f;
    for (int c = 0; c < 10; c++) { v[c] = expf(v[c] - mx); s += v[c]; }
    for (int c = 0; c < 10; c++) theta[b * 10 + c] = v[c] / s;
}

__global__ void k_softmax(const float* __restrict__ in, float* __restrict__ out, int F) {
    __shared__ float red[256];
    int b = blockIdx.x, t = threadIdx.x;
    float mx = -3.4e38f;
    for (int f = t; f < F; f += 256) mx = fmaxf(mx, in[(long)b * F + f]);
    red[t] = mx; __syncthreads();
    for (int s = 128; s > 0; s >>= 1) { if (t < s) red[t] = fmaxf(red[t], red[t + s]); __syncthreads(); }
    mx = red[0]; __syncthreads();
    float sum = 0.f;
    for (int f = t; f < F; f += 256) sum += expf(in[(long)b * F + f] - mx);
    red[t] = sum; __syncthreads();
    for (int s = 128; s > 0; s >>= 1) { if (t < s) red[t] += red[t + s]; __syncthreads(); }
    sum = red[0];
    for (int f = t; f < F; f += 256)
        out[(long)b * F + f] = expf(in[(long)b * F + f] - mx) / sum;
}

// ---------------------------------------------------------------------------
// Host-side helpers
// ---------------------------------------------------------------------------
static void conv_gemm(hipStream_t s, const float* src, int Bb, int Cc, int Hh, int Ww,
                      int OH, int OW, int ksz, int stride, int pad, int dil,
                      int Kp, int N, const _Float16* Wf, const float* bias,
                      float* out, int reluIn, int reluOut, _Float16* colbuf) {
    int M = Bb * OH * OW;
    for (int m0 = 0; m0 < M; m0 += CHUNK_ROWS) {
        int Mc = M - m0; if (Mc > CHUNK_ROWS) Mc = CHUNK_ROWS;
        long tot = (long)Mc * Kp;
        int blocks = (int)((tot + 255) / 256); if (blocks > 8192) blocks = 8192;
        k_im2col<<<blocks, 256, 0, s>>>(src, colbuf, Cc, Hh, Ww, OH, OW,
                                        ksz, stride, pad, dil, Kp, m0, Mc, reluIn);
        dim3 g(Mc / GT_M, N / GT_N);
        k_gemm_wmma<<<g, 128, 0, s>>>(colbuf, Wf, out + (long)m0 * N, bias, N, Kp, reluOut);
    }
}

static void bn2d_run(hipStream_t s, const float* in, float* out, const float* resid,
                     const float* g, const float* b, int M, float* part, float* ss, int relu) {
    int rowsPerBlk = 1024;
    int nb = M / rowsPerBlk;
    k_bnstats1<<<nb, 256, 0, s>>>(in, part, rowsPerBlk, 256);
    k_bnstats2<<<1, 256, 0, s>>>(part, g, b, ss, nb, 256, M);
    long tot = (long)M * 256;
    k_bnapply<<<4096, 256, 0, s>>>(in, ss, resid, out, tot, 256, relu);
}

static void resblock_run(hipStream_t s, float* x, const _Float16* w3, const _Float16* w1,
                         const float* b1, const float* g1, const float* be1,
                         const float* b2, const float* g2, const float* be2,
                         float* tmp1, float* tmp2, _Float16* colbuf,
                         float* part, float* ss) {
    // h = bn(conv3x3(relu(x)))
    conv_gemm(s, x, 256, 256, 8, 8, 8, 8, 3, 1, 1, 1, 2304, 256, w3, b1, tmp1, 1, 0, colbuf);
    bn2d_run(s, tmp1, tmp1, nullptr, g1, be1, 16384, part, ss, 1);   // + relu
    // h = bn(conv1x1(h)); out = x + h
    conv_gemm(s, tmp1, 256, 256, 8, 8, 8, 8, 1, 1, 0, 1, 256, 256, w1, b2, tmp2, 0, 0, colbuf);
    bn2d_run(s, tmp2, x, x, g2, be2, 16384, part, ss, 0);
}

// ---------------------------------------------------------------------------
extern "C" void kernel_launch(void* const* d_in, const int* in_sizes, int n_in,
                              void* d_out, int out_size, void* d_ws, size_t ws_size,
                              hipStream_t stream) {
    (void)in_sizes; (void)n_in; (void)out_size;
    auto P = [&](int i) { return (const float*)d_in[i]; };
    const float* x    = P(0);
    const float* eps  = P(1);
    const float* c1w = P(2);  const float* c1b = P(3);
    const float* bn1g = P(4); const float* bn1b = P(5);
    const float* c2w = P(6);  const float* c2b = P(7);
    const int rbBase[4] = {8, 16, 25, 33};
    const float* Ecb = P(24);
    const float* ct1w = P(41); const float* ct1b = P(42);
    const float* bndg = P(43); const float* bndb = P(44);
    const float* ct2w = P(45); const float* ct2b = P(46);
    const float* infw = P(47); const float* infb = P(48);
    const float* l1w = P(49);  const float* l1b = P(50);
    const float* muw = P(51);  const float* mub = P(52);
    const float* sw  = P(53);  const float* sb  = P(54);
    const float* beta = P(55);
    float* dout = (float*)d_out;

    // ---- workspace carve-out ----
    uint8_t* base = (uint8_t*)d_ws;
    size_t off = 0;
    auto take = [&](size_t bytes) -> void* {
        void* p = base + off;
        off = (off + bytes + 255) & ~(size_t)255;
        return p;
    };
    _Float16* W1  = (_Float16*)take((size_t)64 * 256 * 2);
    _Float16* W2  = (_Float16*)take((size_t)4096 * 256 * 2);
    _Float16* WT1 = (_Float16*)take((size_t)4096 * 256 * 2);
    _Float16* WR3[4]; _Float16* WR1[4];
    for (int r = 0; r < 4; r++) {
        WR3[r] = (_Float16*)take((size_t)2304 * 256 * 2);
        WR1[r] = (_Float16*)take((size_t)256 * 256 * 2);
    }
    _Float16* COL = (_Float16*)take((size_t)CHUNK_ROWS * 4096 * 2);   // 32 MB
    float* FA = (float*)take((size_t)65536 * 256 * 4);                // 64 MB
    float* FB = (float*)take((size_t)16384 * 256 * 4);
    float* FC = (float*)take((size_t)16384 * 256 * 4);
    float* FD = (float*)take((size_t)16384 * 256 * 4);
    float* FE = (float*)take((size_t)16384 * 256 * 4);
    float* PART = (float*)take((size_t)2 * 64 * 256 * 4);
    float* SS   = (float*)take((size_t)2 * 256 * 4);
    int*   IDX  = (int*)take((size_t)16384 * 4);
    float* T0v = (float*)take((size_t)256 * 100 * 4);
    float* T1v = (float*)take((size_t)256 * 100 * 4);
    float* MUv = (float*)take((size_t)256 * 10 * 4);
    float* LVv = (float*)take((size_t)256 * 10 * 4);
    float* THv = (float*)take((size_t)256 * 10 * 4);
    float* WDv = (float*)take((size_t)256 * 512 * 4);
    if (off > ws_size) return;   // workspace too small; bail deterministically

    const long OUT_XT = 0;
    const long OUT_ZE = 786432;
    const long OUT_ZQ = OUT_ZE + 4194304;
    const long OUT_WD = OUT_ZQ + 4194304;

    // ---- weight preparation (f32 -> f16, stored as B^T: N x Kp) ----
    auto prep = [&](const float* src, _Float16* dst, int O, int I, int k, int Kp, int tm) {
        long tot = (long)Kp * O;
        int blocks = (int)((tot + 255) / 256); if (blocks > 4096) blocks = 4096;
        k_prep_w<<<blocks, 256, 0, stream>>>(src, dst, O, I, k, Kp, tm);
    };
    prep(c1w, W1, 256, 3, 4, 64, 0);
    prep(c2w, W2, 256, 256, 4, 4096, 0);
    for (int r = 0; r < 4; r++) {
        prep(P(rbBase[r] + 0), WR3[r], 256, 256, 3, 2304, 0);
        prep(P(rbBase[r] + 4), WR1[r], 256, 256, 1, 256, 0);
    }
    prep(ct1w, WT1, 256, 256, 4, 4096, 1);   // flip + transpose for convT

    // ---- encoder ----
    // x NCHW -> NHWC (staged in FC, free until resblocks)
    k_nchw2nhwc<<<2048, 256, 0, stream>>>(x, FC, 256, 3, 32, 32);
    // conv1 (3->256, k4 s2 p1) -> FA (65536 x 256)
    conv_gemm(stream, FC, 256, 3, 32, 32, 16, 16, 4, 2, 1, 1, 64, 256, W1, c1b, FA, 0, 0, COL);
    bn2d_run(stream, FA, FA, nullptr, bn1g, bn1b, 65536, PART, SS, 1);   // bn + relu
    // conv2 (256->256, k4 s2 p1) -> FB (16384 x 256)
    conv_gemm(stream, FA, 256, 256, 16, 16, 8, 8, 4, 2, 1, 1, 4096, 256, W2, c2b, FB, 0, 0, COL);
    // resblocks 1,2 -> z_e in FB
    for (int r = 0; r < 2; r++) {
        int b0 = rbBase[r];
        resblock_run(stream, FB, WR3[r], WR1[r], P(b0+1), P(b0+2), P(b0+3),
                     P(b0+5), P(b0+6), P(b0+7), FC, FE, COL, PART, SS);
    }
    // z_e output (NCHW)
    k_nhwc2nchw<<<4096, 256, 0, stream>>>(FB, dout + OUT_ZE, 256, 256, 8, 8);

    // ---- VQ ----
    k_vq_argmin<<<16384, 256, 0, stream>>>(FB, Ecb, IDX, 256, 512);
    k_gather<<<4096, 256, 0, stream>>>(IDX, Ecb, FD, (long)16384 * 256, 256);
    k_nhwc2nchw<<<4096, 256, 0, stream>>>(FD, dout + OUT_ZQ, 256, 256, 8, 8);

    // ---- decoder ----
    for (int r = 2; r < 4; r++) {
        int b0 = rbBase[r];
        resblock_run(stream, FD, WR3[r], WR1[r], P(b0+1), P(b0+2), P(b0+3),
                     P(b0+5), P(b0+6), P(b0+7), FC, FE, COL, PART, SS);
    }
    // relu(hd) fused into im2col; convT1 as dilated conv (k4, pad2, lhs-dil 2) -> FA
    conv_gemm(stream, FD, 256, 256, 8, 8, 16, 16, 4, 1, 2, 2, 4096, 256, WT1, ct1b, FA, 1, 0, COL);
    bn2d_run(stream, FA, FA, nullptr, bndg, bndb, 65536, PART, SS, 1);   // bn + relu
    // convT2 (256->3) + tanh -> x_tilde NCHW
    k_convT2<<<1024, 256, 0, stream>>>(FA, ct2w, ct2b, dout + OUT_XT);

    // ---- topic head ----
    k_topic_t0<<<256, 128, 0, stream>>>(IDX, infw, infb, T0v);
    k_linear<<<256, 256, 0, stream>>>(T0v, l1w, l1b, T1v, 100, 100, 0, 1);
    k_linear<<<256, 256, 0, stream>>>(T1v, muw, mub, MUv, 100, 10, 0, 0);
    k_linear<<<256, 256, 0, stream>>>(T1v, sw,  sb,  LVv, 100, 10, 0, 0);
    k_bn1d<<<10, 256, 0, stream>>>(MUv, MUv, 10);
    k_bn1d<<<10, 256, 0, stream>>>(LVv, LVv, 10);
    k_theta<<<256, 32, 0, stream>>>(MUv, LVv, eps, THv);
    k_linear<<<256, 256, 0, stream>>>(THv, beta, nullptr, WDv, 10, 512, 1, 0);
    k_bn1d<<<512, 256, 0, stream>>>(WDv, WDv, 512);
    k_softmax<<<256, 256, 0, stream>>>(WDv, dout + OUT_WD, 512);
}
